// AFRM_48636209660262
// MI455X (gfx1250) — compile-verified, matrix operand order checked by reading, not run on hardware
//
#include <hip/hip_runtime.h>
#include <hip/hip_bf16.h>
#include <stdint.h>

typedef __attribute__((ext_vector_type(16))) __bf16 bf16x16;
typedef __attribute__((ext_vector_type(8)))  float  f32x8;
typedef __attribute__((ext_vector_type(4)))  unsigned int u32x4;

union Frag { bf16x16 v; u32x4 q[2]; };

__device__ __forceinline__ unsigned short f2bf(float f) {
    unsigned u = __builtin_bit_cast(unsigned, f);
    unsigned r = u + 0x7fffu + ((u >> 16) & 1u);   // round-to-nearest-even
    return (unsigned short)(r >> 16);
}
__device__ __forceinline__ float sigm(float x) { return 1.f / (1.f + __expf(-x)); }

// ---------------------------------------------------------------------------
// Weight packing: fragment-native layout.
// Per (k-chunk c of 32, n-tile t of 16): lane l holds n = 16*t + (l&15),
// 16 bf16 elems e: k = 32*c + 8*(l>>4) + (e<8 ? e : e+8).
// ---------------------------------------------------------------------------
__global__ void pack_rowmajor(const float* __restrict__ src,
                              unsigned short* __restrict__ dst, int K, int N) {
    int tid = blockIdx.x * 256 + threadIdx.x;
    if (tid >= K * N) return;
    int e   = tid & 15;
    int l   = (tid >> 4) & 31;
    int rem = tid >> 9;
    int nt  = N >> 4;
    int t   = rem % nt;
    int c   = rem / nt;
    int k   = (c << 5) + ((l >> 4) << 3) + (e < 8 ? e : e + 8);
    int n   = (t << 4) + (l & 15);
    dst[tid] = f2bf(src[(size_t)k * N + n]);
}

// Decoder weights: per layer, per parity class (py,px), K = 4 taps x 256.
// tap t -> kyIdx=t>>1, kxIdx=t&1; ky=(1-py)+2*kyIdx, kx=(1-px)+2*kxIdx.
__global__ void pack_dec(const float* __restrict__ dec_w,
                         unsigned short* __restrict__ dst) {
    const int C = 256;
    int tid = blockIdx.x * 256 + threadIdx.x;     // 4*4*1024*256
    if (tid >= 4 * 4 * 1024 * 256) return;
    int inner = tid & (1024 * 256 - 1);
    int lc    = tid >> 18;
    int layer = lc >> 2, cls = lc & 3;
    int py = (cls >> 1) & 1, px = cls & 1;
    int e   = inner & 15;
    int l   = (inner >> 4) & 31;
    int rem = inner >> 9;
    int t16 = rem & 15;
    int c   = rem >> 4;
    int k   = (c << 5) + ((l >> 4) << 3) + (e < 8 ? e : e + 8);
    int n   = (t16 << 4) + (l & 15);
    int tap = k >> 8;
    int ci  = k & 255;
    int ky  = (1 - py) + 2 * (tap >> 1);
    int kx  = (1 - px) + 2 * (tap & 1);
    size_t srcoff = ((((size_t)layer * 16) + (size_t)(ky * 4 + kx)) * C + ci) * C + n;
    dst[tid] = f2bf(dec_w[srcoff]);
}

// Zero the 1-pixel halo of a padded [B][H+2][H+2][256] bf16 buffer.
__global__ void zero_halo(unsigned short* __restrict__ buf, int H) {
    int Hp = H + 2;
    int local = blockIdx.x;       // 0 .. 4H+3
    int b = blockIdx.y;
    int y, xx;
    if (local < Hp)            { y = 0;      xx = local; }
    else if (local < 2 * Hp)   { y = Hp - 1; xx = local - Hp; }
    else { int rem = local - 2 * Hp; y = (rem >> 1) + 1; xx = (rem & 1) ? Hp - 1 : 0; }
    buf[(((size_t)b * Hp + y) * Hp + xx) * 256 + threadIdx.x] = 0;
}

// x [B,64,64,256] f32 -> padded bf16 [B,66,66,256] interior
__global__ void cvt_pad(const float* __restrict__ src,
                        unsigned short* __restrict__ dst) {
    int tid = blockIdx.x * 256 + threadIdx.x;   // 33,554,432
    int ch = tid & 255; int xx = (tid >> 8) & 63; int y = (tid >> 14) & 63; int b = tid >> 20;
    dst[(((size_t)b * 66 + y + 1) * 66 + xx + 1) * 256 + ch] = f2bf(src[tid]);
}

// ---------------------------------------------------------------------------
// Encoder conv: 4x4 s2 SAME via implicit GEMM on padded input (branch-free).
// 8 waves (2Mg x 4N); wave tile 32M x 64N; A double-buffered, B rolling.
// ---------------------------------------------------------------------------
template <int Hin>
__global__ __launch_bounds__(256, 2)
void enc_conv(const unsigned short* __restrict__ act,
              const unsigned short* __restrict__ wpack,
              const float* __restrict__ bias, const float* __restrict__ bng,
              const float* __restrict__ bnb,  const float* __restrict__ bnm,
              const float* __restrict__ bnv,
              unsigned short* __restrict__ out) {
    const int C   = 256;
    const int Ho  = Hin >> 1;
    const int Hp  = Hin + 2;       // padded input
    const int Hop = Ho + 2;        // padded output
    const int lane = threadIdx.x & 31;
    const int wave = threadIdx.x >> 5;
    const int wm = wave >> 2, wn = wave & 3;
    const int kbase = (lane >> 4) << 3;
    const int HoWo = Ho * Ho;

    int ofs[2];
#pragma unroll
    for (int r = 0; r < 2; ++r) {
        int m  = blockIdx.x * 64 + wm * 32 + r * 16 + (lane & 15);
        int b  = m / HoWo;
        int r0 = m - b * HoWo;
        int oy = r0 / Ho;
        int ox = r0 - oy * Ho;
        // base at padded (oy*2, ox*2) == unpadded (oy*2-1, ox*2-1) + halo
        ofs[r] = ((b * Hp + oy * 2) * Hp + ox * 2) * C + kbase;
    }

    const unsigned short* wbase = wpack + ((size_t)(wn * 4) * 32 + lane) * 16;

    f32x8 acc[2][4];
    const f32x8 z8 = {0.f,0.f,0.f,0.f,0.f,0.f,0.f,0.f};
#pragma unroll
    for (int r = 0; r < 2; ++r)
#pragma unroll
        for (int j = 0; j < 4; ++j) acc[r][j] = z8;

    Frag afr[2][2], bfr[2];

    auto loadA = [&](int buf, int c) {
        int tap = c >> 3;
        int ci0 = (c & 7) << 5;
        int ky = tap >> 2, kx = tap & 3;
        int add = (ky * Hp + kx) * C + ci0;
#pragma unroll
        for (int r = 0; r < 2; ++r) {
            const u32x4* p = (const u32x4*)(act + (ofs[r] + add));
            afr[buf][r].q[0] = p[0]; afr[buf][r].q[1] = p[2];
        }
    };
    auto loadBf = [&](int buf, int f) {
        const u32x4* wp = (const u32x4*)(wbase + (size_t)(f >> 2) * 8192
                                               + (size_t)(f & 3) * 512);
        bfr[buf].q[0] = wp[0]; bfr[buf].q[1] = wp[1];
    };

    const int KC = 128;   // K = 4096
    loadA(0, 0); loadBf(0, 0);
    for (int c = 0; c < KC; ++c) {
        int curA = c & 1;
        if (c + 1 < KC) loadA(curA ^ 1, c + 1);
#pragma unroll
        for (int j = 0; j < 4; ++j) {
            int f = c * 4 + j;
            if (f + 1 < KC * 4) loadBf((j + 1) & 1, f + 1);
#pragma unroll
            for (int r = 0; r < 2; ++r)
                acc[r][j] = __builtin_amdgcn_wmma_f32_16x16x32_bf16(
                    false, afr[curA][r].v, false, bfr[j & 1].v, (short)0,
                    acc[r][j], false, false);
        }
    }

#pragma unroll
    for (int r = 0; r < 2; ++r) {
        const int mv0 = blockIdx.x * 64 + wm * 32 + r * 16 + ((lane >> 4) << 3);
#pragma unroll
        for (int j = 0; j < 4; ++j) {
            int n = wn * 64 + j * 16 + (lane & 15);
            float bs  = bias[n];
            float g   = bng[n], bb = bnb[n], mm = bnm[n];
            float inv = rsqrtf(bnv[n] + 1e-3f);
#pragma unroll
            for (int v = 0; v < 8; ++v) {
                int mrow = mv0 + v;
                int b2 = mrow / HoWo;
                int rr = mrow - b2 * HoWo;
                int oy = rr / Ho;
                int ox = rr - oy * Ho;
                float t = acc[r][j][v] + bs;
                t = (t - mm) * inv * g + bb;
                t = t > 0.f ? t : 0.f;
                out[(((size_t)b2 * Hop + oy + 1) * Hop + ox + 1) * C + n] = f2bf(t);
            }
        }
    }
}

// ---------------------------------------------------------------------------
// Decoder conv_transpose on padded input.  Parity class = blockIdx.y.
// class K = 1024.  FINAL fuses gamma*h + x -> f32 (unpadded out).
// ---------------------------------------------------------------------------
template <int Hin, int FINAL>
__global__ __launch_bounds__(256, 2)
void dec_conv(const unsigned short* __restrict__ act,
              const unsigned short* __restrict__ wpack4,
              const float* __restrict__ bias, const float* __restrict__ bng,
              const float* __restrict__ bnb,  const float* __restrict__ bnm,
              const float* __restrict__ bnv,
              unsigned short* __restrict__ out_bf, float* __restrict__ out_f32,
              const float* __restrict__ xres, const float* __restrict__ gamma) {
    const int C   = 256;
    const int Ho  = Hin << 1;
    const int Hp  = Hin + 2;
    const int Hop = Ho + 2;
    const int cls = blockIdx.y;
    const int py = (cls >> 1) & 1, px = cls & 1;
    const unsigned short* wpack = wpack4 + (size_t)cls * 1024 * 256;

    const int lane = threadIdx.x & 31;
    const int wave = threadIdx.x >> 5;
    const int wm = wave >> 2, wn = wave & 3;
    const int kbase = (lane >> 4) << 3;
    const int HW = Hin * Hin;

    int ofs[2];
#pragma unroll
    for (int r = 0; r < 2; ++r) {
        int m   = blockIdx.x * 64 + wm * 32 + r * 16 + (lane & 15);
        int b   = m / HW;
        int r0  = m - b * HW;
        int oy2 = r0 / Hin;
        int ox2 = r0 - oy2 * Hin;
        // tap kyIdx=0 reads iy0 = oy2+py; padded row iy0+1
        ofs[r] = ((b * Hp + oy2 + py + 1) * Hp + ox2 + px + 1) * C + kbase;
    }

    const unsigned short* wbase = wpack + ((size_t)(wn * 4) * 32 + lane) * 16;

    f32x8 acc[2][4];
    const f32x8 z8 = {0.f,0.f,0.f,0.f,0.f,0.f,0.f,0.f};
#pragma unroll
    for (int r = 0; r < 2; ++r)
#pragma unroll
        for (int j = 0; j < 4; ++j) acc[r][j] = z8;

    Frag afr[2][2], bfr[2];

    auto loadA = [&](int buf, int c) {
        int tap = c >> 3;
        int ci0 = (c & 7) << 5;
        int kyI = tap >> 1, kxI = tap & 1;
        int add = ci0 - (kyI * Hp + kxI) * C;
#pragma unroll
        for (int r = 0; r < 2; ++r) {
            const u32x4* p = (const u32x4*)(act + (ofs[r] + add));
            afr[buf][r].q[0] = p[0]; afr[buf][r].q[1] = p[2];
        }
    };
    auto loadBf = [&](int buf, int f) {
        const u32x4* wp = (const u32x4*)(wbase + (size_t)(f >> 2) * 8192
                                               + (size_t)(f & 3) * 512);
        bfr[buf].q[0] = wp[0]; bfr[buf].q[1] = wp[1];
    };

    const int KC = 32;              // K = 1024
    loadA(0, 0); loadBf(0, 0);
    for (int c = 0; c < KC; ++c) {
        int curA = c & 1;
        if (c + 1 < KC) loadA(curA ^ 1, c + 1);
#pragma unroll
        for (int j = 0; j < 4; ++j) {
            int f = c * 4 + j;
            if (f + 1 < KC * 4) loadBf((j + 1) & 1, f + 1);
#pragma unroll
            for (int r = 0; r < 2; ++r)
                acc[r][j] = __builtin_amdgcn_wmma_f32_16x16x32_bf16(
                    false, afr[curA][r].v, false, bfr[j & 1].v, (short)0,
                    acc[r][j], false, false);
        }
    }

    const float gma = FINAL ? gamma[0] : 0.f;
#pragma unroll
    for (int r = 0; r < 2; ++r) {
        const int mv0 = blockIdx.x * 64 + wm * 32 + r * 16 + ((lane >> 4) << 3);
#pragma unroll
        for (int j = 0; j < 4; ++j) {
            int n = wn * 64 + j * 16 + (lane & 15);
            float bs  = bias[n];
            float g   = bng[n], bb = bnb[n], mm = bnm[n];
            float inv = rsqrtf(bnv[n] + 1e-3f);
#pragma unroll
            for (int v = 0; v < 8; ++v) {
                int mrow = mv0 + v;
                int b2   = mrow / HW;
                int rr   = mrow - b2 * HW;
                int oy2r = rr / Hin;
                int ox2r = rr - oy2r * Hin;
                int oyr = 2 * oy2r + py, oxr = 2 * ox2r + px;
                float t = acc[r][j][v] + bs;
                t = (t - mm) * inv * g + bb;
                t = t > 0.f ? t : 0.f;
                if (FINAL) {
                    size_t idx = (((size_t)b2 * Ho + oyr) * Ho + oxr) * C + n;
                    out_f32[idx] = gma * t + xres[idx];
                } else {
                    size_t idx = (((size_t)b2 * Hop + oyr + 1) * Hop + oxr + 1) * C + n;
                    out_bf[idx] = f2bf(t);
                }
            }
        }
    }
}

// ---------------------------------------------------------------------------
// Generic GEMM: A [M,K] bf16 row-major x packed B [K,N]. Wave tile 16Mx64N,
// A double-buffered, B rolling.  MODE: 0 Cf32=acc+bias; 1 Cf32+=acc;
// 2 Cbf16=leaky(acc+bias); 3 like 2 but store padded [B][6][6][256] (H=4).
// ---------------------------------------------------------------------------
template <int K, int N, int MODE>
__global__ __launch_bounds__(256, 2)
void gemm_bf16(const unsigned short* __restrict__ A,
               const unsigned short* __restrict__ wpack,
               const float* __restrict__ bias, float* __restrict__ Cf32,
               unsigned short* __restrict__ Cbf16) {
    const int lane = threadIdx.x & 31;
    const int wave = threadIdx.x >> 5;
    const int wm = wave >> 2, wn = wave & 3;
    const int m  = blockIdx.x * 32 + wm * 16 + (lane & 15);
    const int kbase = (lane >> 4) << 3;
    const int NT = N >> 4;

    const unsigned short* abase = A + (size_t)m * K + kbase;
    const unsigned short* wbase = wpack +
        ((size_t)(blockIdx.y * 16 + wn * 4) * 32 + lane) * 16;

    f32x8 acc[4];
    const f32x8 z8 = {0.f,0.f,0.f,0.f,0.f,0.f,0.f,0.f};
#pragma unroll
    for (int j = 0; j < 4; ++j) acc[j] = z8;

    Frag afr[2], bfr[2];
    auto loadA = [&](int buf, int c) {
        const u32x4* p = (const u32x4*)(abase + (c << 5));
        afr[buf].q[0] = p[0]; afr[buf].q[1] = p[2];
    };
    auto loadBf = [&](int buf, int f) {
        const u32x4* wp = (const u32x4*)(wbase + (size_t)(f >> 2) * (NT * 512)
                                               + (size_t)(f & 3) * 512);
        bfr[buf].q[0] = wp[0]; bfr[buf].q[1] = wp[1];
    };

    const int KC = K >> 5;
    loadA(0, 0); loadBf(0, 0);
    for (int c = 0; c < KC; ++c) {
        int curA = c & 1;
        if (c + 1 < KC) loadA(curA ^ 1, c + 1);
#pragma unroll
        for (int j = 0; j < 4; ++j) {
            int f = c * 4 + j;
            if (f + 1 < KC * 4) loadBf((j + 1) & 1, f + 1);
            acc[j] = __builtin_amdgcn_wmma_f32_16x16x32_bf16(
                false, afr[curA].v, false, bfr[j & 1].v, (short)0, acc[j], false, false);
        }
    }

    const int mv0 = blockIdx.x * 32 + wm * 16 + ((lane >> 4) << 3);
#pragma unroll
    for (int j = 0; j < 4; ++j) {
        int n = blockIdx.y * 256 + wn * 64 + j * 16 + (lane & 15);
        float bs = (MODE != 1) ? bias[n] : 0.f;
#pragma unroll
        for (int v = 0; v < 8; ++v) {
            int mrow = mv0 + v;
            if (MODE == 0) {
                Cf32[(size_t)mrow * N + n] = acc[j][v] + bs;
            } else if (MODE == 1) {
                Cf32[(size_t)mrow * N + n] += acc[j][v];
            } else if (MODE == 2) {
                float t = acc[j][v] + bs;
                t = t > 0.f ? t : 0.3f * t;
                Cbf16[(size_t)mrow * N + n] = f2bf(t);
            } else {
                // mrow = (b*4+hh)*4+ww -> padded [b][hh+1][ww+1]
                int ww = mrow & 3, hh = (mrow >> 2) & 3, b = mrow >> 4;
                float t = acc[j][v] + bs;
                t = t > 0.f ? t : 0.3f * t;
                Cbf16[(((size_t)b * 6 + hh + 1) * 6 + ww + 1) * 256 + n] = f2bf(t);
            }
        }
    }
}

// seq[s][b][(hh*4+ww)*64+cc] = act4_padded[b][hh+1][ww+1][s*64+cc]
__global__ void build_seq(const unsigned short* __restrict__ act4,
                          unsigned short* __restrict__ seq) {
    int tid = blockIdx.x * 256 + threadIdx.x;   // 4*32*1024
    int l = tid & 1023; int bb = (tid >> 10) & 31; int s = tid >> 15;
    int hh = l >> 8; int ww = (l >> 6) & 3; int cc = l & 63;
    seq[tid] = act4[((((size_t)bb * 6 + hh + 1) * 6 + ww + 1) << 8) + s * 64 + cc];
}

// cat[m][k]: m=(b*4+hh)*4+ww; k<256: hsf[ww][b][hh*256+k] else hsr same
__global__ void build_cat(const unsigned short* __restrict__ hsf,
                          const unsigned short* __restrict__ hsr,
                          unsigned short* __restrict__ cat) {
    int tid = blockIdx.x * 256 + threadIdx.x;   // 512*512
    int k = tid & 511; int m = tid >> 9;
    int ww = m & 3; int hh = (m >> 2) & 3; int b = m >> 4;
    const unsigned short* src = (k < 256) ? hsf : hsr;
    int kk = k & 255;
    cat[tid] = src[((ww * 32 + b) << 10) + (hh << 8) + kk];
}

__global__ void lstm_cell(const float* __restrict__ z, float* __restrict__ cbuf,
                          unsigned short* __restrict__ hprev,
                          unsigned short* __restrict__ hs, int slot, int first) {
    int tid = blockIdx.x * 256 + threadIdx.x;   // 32*1024
    int b = tid >> 10; int l = tid & 1023;
    const float* zb = z + ((size_t)slot * 32 + b) * 4096;
    float zi = zb[l], zf = zb[l + 1024], zg = zb[l + 2048], zo = zb[l + 3072];
    float cp = first ? 0.f : cbuf[tid];
    float cn = sigm(zf) * cp + sigm(zi) * tanhf(zg);
    float h  = sigm(zo) * tanhf(cn);
    cbuf[tid] = cn;
    unsigned short hb = f2bf(h);
    hprev[tid] = hb;
    hs[(size_t)slot * 32768 + tid] = hb;
}

// ---------------------------------------------------------------------------
extern "C" void kernel_launch(void* const* d_in, const int* in_sizes, int n_in,
                              void* d_out, int out_size, void* d_ws, size_t ws_size,
                              hipStream_t stream) {
    (void)in_sizes; (void)n_in; (void)out_size; (void)ws_size;
    const float* x        = (const float*)d_in[0];
    const float* enc_w    = (const float*)d_in[1];
    const float* enc_b    = (const float*)d_in[2];
    const float* enc_g    = (const float*)d_in[3];
    const float* enc_bb   = (const float*)d_in[4];
    const float* enc_m    = (const float*)d_in[5];
    const float* enc_v    = (const float*)d_in[6];
    const float* dec_w    = (const float*)d_in[7];
    const float* dec_b    = (const float*)d_in[8];
    const float* dec_g    = (const float*)d_in[9];
    const float* dec_bb   = (const float*)d_in[10];
    const float* dec_m    = (const float*)d_in[11];
    const float* dec_v    = (const float*)d_in[12];
    const float* Wf       = (const float*)d_in[13];
    const float* Uf       = (const float*)d_in[14];
    const float* bf_      = (const float*)d_in[15];
    const float* Wr       = (const float*)d_in[16];
    const float* Ur       = (const float*)d_in[17];
    const float* br_      = (const float*)d_in[18];
    const float* ffw      = (const float*)d_in[19];
    const float* ffb      = (const float*)d_in[20];
    const float* gamma    = (const float*)d_in[21];
    float* out = (float*)d_out;

    char* ws = (char*)d_ws;
    size_t off = 0;
    auto alloc = [&](size_t bytes) { size_t r = off; off += (bytes + 255) & ~(size_t)255; return r; };
    unsigned short* actA = (unsigned short*)(ws + alloc((size_t)32*66*66*256*2));
    unsigned short* actB = (unsigned short*)(ws + alloc((size_t)32*34*34*256*2));
    unsigned short* wenc = (unsigned short*)(ws + alloc((size_t)4*4096*256*2));
    unsigned short* wdec = (unsigned short*)(ws + alloc((size_t)16*1024*256*2));
    unsigned short* wWf  = (unsigned short*)(ws + alloc((size_t)1024*4096*2));
    unsigned short* wUf  = (unsigned short*)(ws + alloc((size_t)1024*4096*2));
    unsigned short* wWr  = (unsigned short*)(ws + alloc((size_t)1024*4096*2));
    unsigned short* wUr  = (unsigned short*)(ws + alloc((size_t)1024*4096*2));
    unsigned short* wff  = (unsigned short*)(ws + alloc((size_t)512*256*2));
    float* zf            = (float*)(ws + alloc((size_t)4*32*4096*4));
    float* zr            = (float*)(ws + alloc((size_t)4*32*4096*4));
    float* cf            = (float*)(ws + alloc((size_t)32*1024*4));
    float* cr            = (float*)(ws + alloc((size_t)32*1024*4));
    unsigned short* hpf  = (unsigned short*)(ws + alloc((size_t)32*1024*2));
    unsigned short* hpr  = (unsigned short*)(ws + alloc((size_t)32*1024*2));
    unsigned short* hsf  = (unsigned short*)(ws + alloc((size_t)4*32*1024*2));
    unsigned short* hsr  = (unsigned short*)(ws + alloc((size_t)4*32*1024*2));
    unsigned short* seqb = (unsigned short*)(ws + alloc((size_t)128*1024*2));
    unsigned short* catb = (unsigned short*)(ws + alloc((size_t)512*512*2));

    // --- pack weights ---
    for (int i = 0; i < 4; ++i)
        pack_rowmajor<<<4096, 256, 0, stream>>>(enc_w + (size_t)i*4096*256,
                                                wenc + (size_t)i*4096*256, 4096, 256);
    pack_dec<<<16384, 256, 0, stream>>>(dec_w, wdec);
    pack_rowmajor<<<16384, 256, 0, stream>>>(Wf, wWf, 1024, 4096);
    pack_rowmajor<<<16384, 256, 0, stream>>>(Uf, wUf, 1024, 4096);
    pack_rowmajor<<<16384, 256, 0, stream>>>(Wr, wWr, 1024, 4096);
    pack_rowmajor<<<16384, 256, 0, stream>>>(Ur, wUr, 1024, 4096);
    pack_rowmajor<<<512,   256, 0, stream>>>(ffw, wff, 512, 256);

    // --- x -> padded bf16 ---
    zero_halo<<<dim3(260, 32), 256, 0, stream>>>(actA, 64);
    cvt_pad<<<131072, 256, 0, stream>>>(x, actA);

    // --- encoder ---
    zero_halo<<<dim3(132, 32), 256, 0, stream>>>(actB, 32);
    enc_conv<64><<<512, 256, 0, stream>>>(actA, wenc + 0*4096*256, enc_b + 0,
        enc_g + 0, enc_bb + 0, enc_m + 0, enc_v + 0, actB);
    zero_halo<<<dim3(68, 32), 256, 0, stream>>>(actA, 16);
    enc_conv<32><<<128, 256, 0, stream>>>(actB, wenc + (size_t)1*4096*256, enc_b + 256,
        enc_g + 256, enc_bb + 256, enc_m + 256, enc_v + 256, actA);
    zero_halo<<<dim3(36, 32), 256, 0, stream>>>(actB, 8);
    enc_conv<16><<<32,  256, 0, stream>>>(actA, wenc + (size_t)2*4096*256, enc_b + 512,
        enc_g + 512, enc_bb + 512, enc_m + 512, enc_v + 512, actB);
    zero_halo<<<dim3(20, 32), 256, 0, stream>>>(actA, 4);
    enc_conv<8><<<8,    256, 0, stream>>>(actB, wenc + (size_t)3*4096*256, enc_b + 768,
        enc_g + 768, enc_bb + 768, enc_m + 768, enc_v + 768, actA);

    // --- LSTM ---
    build_seq<<<512, 256, 0, stream>>>(actA, seqb);
    gemm_bf16<1024,4096,0><<<dim3(4, 16), 256, 0, stream>>>(seqb, wWf, bf_, zf, nullptr);
    gemm_bf16<1024,4096,0><<<dim3(4, 16), 256, 0, stream>>>(seqb, wWr, br_, zr, nullptr);
    for (int t = 0; t < 4; ++t) {
        if (t) gemm_bf16<1024,4096,1><<<dim3(1, 16), 256, 0, stream>>>(
            hpf, wUf, nullptr, zf + (size_t)t*131072, nullptr);
        lstm_cell<<<128, 256, 0, stream>>>(zf, cf, hpf, hsf, t, t == 0);
        int slot = 3 - t;
        if (t) gemm_bf16<1024,4096,1><<<dim3(1, 16), 256, 0, stream>>>(
            hpr, wUr, nullptr, zr + (size_t)slot*131072, nullptr);
        lstm_cell<<<128, 256, 0, stream>>>(zr, cr, hpr, hsr, slot, t == 0);
    }

    // --- ffwd 1x1 conv (512->256) + LeakyReLU -> padded decoder input ---
    build_cat<<<1024, 256, 0, stream>>>(hsf, hsr, catb);
    zero_halo<<<dim3(20, 32), 256, 0, stream>>>(actB, 4);
    gemm_bf16<512,256,3><<<dim3(16, 1), 256, 0, stream>>>(catb, wff, ffb, nullptr, actB);

    // --- decoder ---
    zero_halo<<<dim3(36, 32), 256, 0, stream>>>(actA, 8);
    dec_conv<4,0><<<dim3(8, 4),   256, 0, stream>>>(actB, wdec + (size_t)0*4*262144,
        dec_b + 0, dec_g + 0, dec_bb + 0, dec_m + 0, dec_v + 0,
        actA, nullptr, nullptr, gamma);
    zero_halo<<<dim3(68, 32), 256, 0, stream>>>(actB, 16);
    dec_conv<8,0><<<dim3(32, 4),  256, 0, stream>>>(actA, wdec + (size_t)1*4*262144,
        dec_b + 256, dec_g + 256, dec_bb + 256, dec_m + 256, dec_v + 256,
        actB, nullptr, nullptr, gamma);
    zero_halo<<<dim3(132, 32), 256, 0, stream>>>(actA, 32);
    dec_conv<16,0><<<dim3(128, 4), 256, 0, stream>>>(actB, wdec + (size_t)2*4*262144,
        dec_b + 512, dec_g + 512, dec_bb + 512, dec_m + 512, dec_v + 512,
        actA, nullptr, nullptr, gamma);
    dec_conv<32,1><<<dim3(512, 4), 256, 0, stream>>>(actA, wdec + (size_t)3*4*262144,
        dec_b + 768, dec_g + 768, dec_bb + 768, dec_m + 768, dec_v + 768,
        nullptr, out, x, gamma);
}
